// Decoder_69217692942765
// MI455X (gfx1250) — compile-verified, matrix operand order checked by reading, not run on hardware
//
#include <hip/hip_runtime.h>

// ---------------------------------------------------------------- constants
constexpr int Bsz  = 4;
constexpr int Tsz  = 512;
constexpr int Ssz  = 512;
constexpr int Dsz  = 1024;
constexpr int Fsz  = 4096;
constexpr int Vsz  = 32000;
constexpr int Hsz  = 16;
constexpr int Lsz  = 6;
constexpr int DHsz = 64;          // D / H
constexpr int BTsz = Bsz * Tsz;   // 2048

typedef _Float16 half_t;
typedef __attribute__((ext_vector_type(16))) _Float16 v16h;
typedef __attribute__((ext_vector_type(8)))  _Float16 v8h;
typedef __attribute__((ext_vector_type(8)))  float    v8f;
typedef __attribute__((ext_vector_type(4)))  unsigned int u32x4;
typedef __attribute__((ext_vector_type(8)))  int i32x8;
typedef __attribute__((ext_vector_type(4)))  int i32x4;

// Tensor Data Mover availability (device pass only; host pass parses fallback)
#if defined(__has_builtin)
# if __has_builtin(__builtin_amdgcn_tensor_load_to_lds)
#  define HAVE_TDM 1
# endif
#endif
#ifndef HAVE_TDM
# define HAVE_TDM 0
#endif

// GEMM tiling: 256 threads = 8 wave32s. Block tile 128x128, K-step 32.
// Wave (rp,wc): rp in 0..3 -> 32-row strip, wc in 0..1 -> 64-col half.
// Each wave: 2 A-frags x 4 B-frags = 8 WMMA 16x16x32 per K step.
constexpr int BM  = 128;
constexpr int BN  = 128;
constexpr int BK  = 32;
constexpr int LDA = 40;  // BK + 8 halfs pad -> 80B row stride (16B aligned)
constexpr int LDB = 40;

// ---------------------------------------------------------------- TDM helper
// 2D tile load: 128 rows x 32 halfs, global row stride = strideElems halfs,
// LDS destination padded +16B per 64B row (matches LDA/LDB = 40 halfs).
__device__ __forceinline__ void tdm_load_tile(unsigned lds_byte_off,
                                              const half_t* gsrc,
                                              int strideElems)
{
#if HAVE_TDM
    unsigned long long ga = (unsigned long long)(__SIZE_TYPE__)gsrc;
    u32x4 g0;
    g0[0] = 1u;                                        // count=1, user descriptor
    g0[1] = lds_byte_off;                              // lds_addr
    g0[2] = (unsigned)(ga & 0xFFFFFFFFu);              // global_addr[31:0]
    g0[3] = (unsigned)((ga >> 32) & 0x01FFFFFFu)       // global_addr[56:32]
          | (2u << 30);                                // type = 2 ("image")
    i32x8 g1;
    g1[0] = (1 << 16)        // data_size = 1 -> 2 bytes
          | (1 << 20)        // pad_enable
          | (3 << 22)        // pad_interval: 16 DWORDs (64B) per pad
          | (3 << 25);       // pad_amount: 4 DWORDs (16B)
    g1[1] = (BK << 16);      // tensor_dim0 = 32   (bits 63:48)
    g1[2] = (128 << 16);     // tensor_dim1 = 128  (bits 95:80)
    g1[3] = (BK << 16);      // tile_dim0 = 32     (bits 127:112)
    g1[4] = 128;             // tile_dim1 = 128    (bits 143:128)
    g1[5] = strideElems;     // tensor_dim0_stride[31:0]
    g1[6] = 0;
    g1[7] = 0;
    i32x4 z4 = {0, 0, 0, 0};
# if __clang_major__ >= 23
    i32x8 z8 = {0, 0, 0, 0, 0, 0, 0, 0};
    __builtin_amdgcn_tensor_load_to_lds(g0, g1, z4, z4, z8, 0);
# else
    __builtin_amdgcn_tensor_load_to_lds(g0, g1, z4, z4, 0);
# endif
#else
    (void)lds_byte_off; (void)gsrc; (void)strideElems;
#endif
}

// ---------------------------------------------------------------- GEMM (WMMA)
// A: [batch] M x K row-major f16
// B: transB=0 -> K x N row-major; transB=1 -> N x K row-major
// outMode: 0 = f32 plain, 1 = f16 plain,
//          2 = f16 head-split  (rows b*T+t, cols h*64+d -> [(b*H+h)*T+t]*64+d)
//          3 = f16 head-merge  (batch z=b*H+h, row t, col d -> (b*T+t)*D+h*64+d)
__launch_bounds__(256)
__global__ void gemm_wmma_kernel(const half_t* __restrict__ A,
                                 const half_t* __restrict__ Bmat,
                                 const float*  __restrict__ bias,
                                 void* __restrict__ Cout,
                                 int M, int N, int K,
                                 long sA, long sB, long sC,
                                 int transB, int outMode, int doRelu, float scale)
{
    __shared__ __align__(16) half_t As[BM * LDA];
    __shared__ __align__(16) half_t Bs[BN * LDB];

    const int z = blockIdx.z;
    const half_t* Ab = A    + (long)z * sA;
    const half_t* Bb = Bmat + (long)z * sB;

    const int m0 = blockIdx.y * BM;
    const int n0 = blockIdx.x * BN;

    const int tid  = threadIdx.x;
    const int wave = tid >> 5;
    const int lane = tid & 31;
    const int rp   = wave >> 1;   // 0..3 -> 32-row strip
    const int wc   = wave & 1;    // 0..1 -> 64-col half
    const int lm   = lane & 15;
    const int lk   = lane >> 4;

    v8f acc[2][4] = {};

    bool useTdmA = false, useTdmB = false;
#if HAVE_TDM
    useTdmA = (m0 + BM <= M);
    useTdmB = (transB != 0) && (n0 + BN <= N);
#endif
    const unsigned asOff = (unsigned)(__SIZE_TYPE__)(void*)&As[0];
    const unsigned bsOff = (unsigned)(__SIZE_TYPE__)(void*)&Bs[0];

    const int ksteps = K / BK;    // all K here are multiples of 32
    for (int ks = 0; ks < ksteps; ++ks) {
        const int k0 = ks * BK;

#if HAVE_TDM
        if ((useTdmA || useTdmB) && tid < 32) {
            if (useTdmA) tdm_load_tile(asOff, Ab + (long)m0 * K + k0, K);
            if (useTdmB) tdm_load_tile(bsOff, Bb + (long)n0 * K + k0, K);
            __builtin_amdgcn_s_wait_tensorcnt(0);
        }
#endif
        // ---- manual A staging: 128x32, 2 x 16B loads/thread (clamped, branch-free)
        if (!useTdmA) {
            #pragma unroll
            for (int i = 0; i < 2; ++i) {
                const int idx = tid + i * 256;
                const int row = idx >> 2;
                const int kc  = (idx & 3) << 3;
                int gr = m0 + row;
                if (gr > M - 1) gr = M - 1;          // clamp: garbage rows never stored
                *(uint4*)(&As[row * LDA + kc]) =
                    *(const uint4*)(Ab + (long)gr * K + k0 + kc);
            }
            if (ks + 1 < ksteps)
                __builtin_prefetch((const void*)(Ab + (long)(m0 + (tid >> 2)) * K
                                                 + k0 + BK), 0, 0);
        }
        // ---- manual B staging into Bs[n][k]
        if (!useTdmB) {
            if (transB) {
                #pragma unroll
                for (int i = 0; i < 2; ++i) {
                    const int idx = tid + i * 256;
                    const int n   = idx >> 2;
                    const int kc  = (idx & 3) << 3;
                    int gn = n0 + n;
                    if (gn > N - 1) gn = N - 1;
                    *(uint4*)(&Bs[n * LDB + kc]) =
                        *(const uint4*)(Bb + (long)gn * K + k0 + kc);
                }
            } else {
                #pragma unroll
                for (int i = 0; i < 2; ++i) {
                    const int idx = tid + i * 256;
                    const int k   = idx >> 4;            // 0..31
                    const int nc  = (idx & 15) << 3;     // 0..120
                    int gn = n0 + nc;
                    if (gn + 8 > N) gn = N - 8;          // N is a multiple of 8
                    half_t tmp[8];
                    *(uint4*)tmp = *(const uint4*)(Bb + (long)(k0 + k) * N + gn);
                    #pragma unroll
                    for (int j = 0; j < 8; ++j) Bs[(nc + j) * LDB + k] = tmp[j];
                }
                if (ks + 1 < ksteps)
                    __builtin_prefetch((const void*)(Bb + (long)(k0 + BK + (tid >> 4)) * N
                                                     + n0), 0, 0);
            }
        }
        __syncthreads();

        // ---- fragments + 8 WMMAs
        v16h afrag[2];
        #pragma unroll
        for (int ta = 0; ta < 2; ++ta) {
            const half_t* ap = &As[(rp * 32 + ta * 16 + lm) * LDA + (lk << 3)];
            v8h lo = *(const v8h*)(ap);
            v8h hi = *(const v8h*)(ap + 16);
            #pragma unroll
            for (int i = 0; i < 8; ++i) { afrag[ta][i] = lo[i]; afrag[ta][8 + i] = hi[i]; }
        }
        #pragma unroll
        for (int tb = 0; tb < 4; ++tb) {
            const half_t* bp = &Bs[(wc * 64 + tb * 16 + lm) * LDB + (lk << 3)];
            v8h lo = *(const v8h*)(bp);
            v8h hi = *(const v8h*)(bp + 16);
            v16h bfrag;
            #pragma unroll
            for (int i = 0; i < 8; ++i) { bfrag[i] = lo[i]; bfrag[8 + i] = hi[i]; }
            #pragma unroll
            for (int ta = 0; ta < 2; ++ta) {
                acc[ta][tb] = __builtin_amdgcn_wmma_f32_16x16x32_f16(
                    false, afrag[ta], false, bfrag, (short)0, acc[ta][tb], false, false);
            }
        }
        __syncthreads();
    }

    // ---- epilogue: C tile: VGPR r -> M = r + 8*lk, N = lane%16
    #pragma unroll
    for (int ta = 0; ta < 2; ++ta) {
        #pragma unroll
        for (int tb = 0; tb < 4; ++tb) {
            const int col = n0 + wc * 64 + tb * 16 + lm;
            if (col >= N) continue;
            const float bv = bias ? bias[col] : 0.0f;
            #pragma unroll
            for (int r = 0; r < 8; ++r) {
                const int row = m0 + rp * 32 + ta * 16 + r + (lk << 3);
                if (row >= M) continue;
                float v = acc[ta][tb][r] * scale + bv;
                if (doRelu && v < 0.0f) v = 0.0f;
                switch (outMode) {
                case 0:
                    ((float*)Cout)[(long)z * sC + (long)row * N + col] = v;
                    break;
                case 1:
                    ((half_t*)Cout)[(long)z * sC + (long)row * N + col] = (half_t)v;
                    break;
                case 2: {   // rows = b*T+t over B*T (T==S), cols = h*64+d
                    const int b  = row >> 9;
                    const int t  = row & (Tsz - 1);
                    const int hh = col >> 6;
                    const int dd = col & 63;
                    ((half_t*)Cout)[(((long)(b * Hsz + hh) * Tsz + t) << 6) + dd] = (half_t)v;
                    break;
                }
                case 3: {   // z = b*H+h, row = t, col = d
                    const int b  = z >> 4;
                    const int hh = z & (Hsz - 1);
                    ((half_t*)Cout)[(long)(b * Tsz + row) * Dsz + hh * 64 + col] = (half_t)v;
                    break;
                }
                }
            }
        }
    }
}

// ---------------------------------------------------------------- elementwise / row kernels
__launch_bounds__(256)
__global__ void f32_to_f16_kernel(const float* __restrict__ src, half_t* __restrict__ dst, long n)
{
    long i = (long)blockIdx.x * 256 + threadIdx.x;
    const long stride = (long)gridDim.x * 256;
    for (; i < n; i += stride) dst[i] = (half_t)src[i];
}

__launch_bounds__(256)
__global__ void embed_kernel(const int* __restrict__ ids, const float* __restrict__ emb,
                             const float* __restrict__ pos,
                             float* __restrict__ xf, half_t* __restrict__ xh)
{
    const int row = blockIdx.x;              // b*T + t
    const int t   = row & (Tsz - 1);
    const long eb = (long)ids[row] * Dsz;
    const long pb = (long)t * Dsz;
    const long ob = (long)row * Dsz;
    #pragma unroll
    for (int i = 0; i < 4; ++i) {
        const int c = threadIdx.x + i * 256;
        const float v = emb[eb + c] * 32.0f + pos[pb + c];   // sqrt(D)=32
        xf[ob + c] = v;
        xh[ob + c] = (half_t)v;
    }
}

__launch_bounds__(256)
__global__ void softmax_kernel(const float* __restrict__ S, half_t* __restrict__ P, int causal)
{
    __shared__ float red[256];
    const int row   = blockIdx.x;            // (b*H+h)*T + tq
    const int tq    = row & (Tsz - 1);
    const long base = (long)row * Ssz;
    const int limit = causal ? (tq + 1) : Ssz;
    const int tid   = threadIdx.x;

    float vals[2];
    float m = -3.4e38f;
    #pragma unroll
    for (int i = 0; i < 2; ++i) {
        const int tk = tid + i * 256;
        const float v = (tk < limit) ? S[base + tk] : -3.4e38f;
        vals[i] = v;
        m = fmaxf(m, v);
    }
    red[tid] = m; __syncthreads();
    for (int st = 128; st > 0; st >>= 1) {
        if (tid < st) red[tid] = fmaxf(red[tid], red[tid + st]);
        __syncthreads();
    }
    m = red[0]; __syncthreads();

    float e[2], sum = 0.0f;
    #pragma unroll
    for (int i = 0; i < 2; ++i) {
        const int tk = tid + i * 256;
        e[i] = (tk < limit) ? __expf(vals[i] - m) : 0.0f;
        sum += e[i];
    }
    red[tid] = sum; __syncthreads();
    for (int st = 128; st > 0; st >>= 1) {
        if (tid < st) red[tid] += red[tid + st];
        __syncthreads();
    }
    const float inv = 1.0f / red[0];
    #pragma unroll
    for (int i = 0; i < 2; ++i) {
        const int tk = tid + i * 256;
        P[base + tk] = (half_t)(e[i] * inv);
    }
}

__launch_bounds__(256)
__global__ void add_ln_kernel(const float* __restrict__ X, const float* __restrict__ Hd,
                              const float* __restrict__ g, const float* __restrict__ b,
                              float* __restrict__ Xo, half_t* __restrict__ Xh)
{
    __shared__ float red[256];
    const long base = (long)blockIdx.x * Dsz;
    const int tid = threadIdx.x;

    float v[4];
    float s = 0.0f;
    #pragma unroll
    for (int i = 0; i < 4; ++i) {
        const int c = tid + i * 256;
        v[i] = X[base + c] + Hd[base + c];
        s += v[i];
    }
    red[tid] = s; __syncthreads();
    for (int st = 128; st > 0; st >>= 1) {
        if (tid < st) red[tid] += red[tid + st];
        __syncthreads();
    }
    const float mean = red[0] * (1.0f / Dsz);
    __syncthreads();

    s = 0.0f;
    #pragma unroll
    for (int i = 0; i < 4; ++i) { const float d = v[i] - mean; s += d * d; }
    red[tid] = s; __syncthreads();
    for (int st = 128; st > 0; st >>= 1) {
        if (tid < st) red[tid] += red[tid + st];
        __syncthreads();
    }
    const float rstd = rsqrtf(red[0] * (1.0f / Dsz) + 1e-5f);

    #pragma unroll
    for (int i = 0; i < 4; ++i) {
        const int c = tid + i * 256;
        const float o = (v[i] - mean) * rstd * g[c] + b[c];
        Xo[base + c] = o;
        Xh[base + c] = (half_t)o;
    }
}

__launch_bounds__(256)
__global__ void log_softmax_kernel(float* __restrict__ logits)
{
    __shared__ float red[256];
    const long base = (long)blockIdx.x * Vsz;
    const int tid = threadIdx.x;

    float m = -3.4e38f;
    for (int c = tid; c < Vsz; c += 256) m = fmaxf(m, logits[base + c]);
    red[tid] = m; __syncthreads();
    for (int st = 128; st > 0; st >>= 1) {
        if (tid < st) red[tid] = fmaxf(red[tid], red[tid + st]);
        __syncthreads();
    }
    m = red[0]; __syncthreads();

    float s = 0.0f;
    for (int c = tid; c < Vsz; c += 256) s += __expf(logits[base + c] - m);
    red[tid] = s; __syncthreads();
    for (int st = 128; st > 0; st >>= 1) {
        if (tid < st) red[tid] += red[tid + st];
        __syncthreads();
    }
    const float lse = m + __logf(red[0]);

    for (int c = tid; c < Vsz; c += 256) logits[base + c] -= lse;
}

// ---------------------------------------------------------------- host orchestration
extern "C" void kernel_launch(void* const* d_in, const int* in_sizes, int n_in,
                              void* d_out, int out_size, void* d_ws, size_t ws_size,
                              hipStream_t stream)
{
    (void)in_sizes; (void)n_in; (void)out_size; (void)ws_size;

    // -------- inputs (setup_inputs dict order; params dict flattened in insertion order)
    const int*   in_ids = (const int*)  d_in[0];
    const float* enc    = (const float*)d_in[1];
    const float* pos    = (const float*)d_in[2];
    // d_in[3] self_attn_mask, d_in[4] srctgt_mask: handled analytically (causal / none)
    int p = 5;
    const float* emb    = (const float*)d_in[p++];
    const float* out_w  = (const float*)d_in[p++];
    const float* ln_g   = (const float*)d_in[p++];
    const float* ln_b   = (const float*)d_in[p++];
    const float* ffn1_w = (const float*)d_in[p++];
    const float* ffn1_b = (const float*)d_in[p++];
    const float* ffn2_w = (const float*)d_in[p++];
    const float* ffn2_b = (const float*)d_in[p++];
    const float *sa_w[4], *sa_b[4], *ca_w[4], *ca_b[4];
    for (int i = 0; i < 4; ++i) { sa_w[i] = (const float*)d_in[p++]; sa_b[i] = (const float*)d_in[p++]; }
    for (int i = 0; i < 4; ++i) { ca_w[i] = (const float*)d_in[p++]; ca_b[i] = (const float*)d_in[p++]; }

    // -------- workspace carve (~225 MB total)
    char* wsb = (char*)d_ws;
    size_t off = 0;
    auto take = [&](size_t bytes) -> void* {
        void* r = wsb + off;
        off = (off + bytes + 255) & ~(size_t)255;
        return r;
    };
    float*  x_f   = (float*) take((size_t)BTsz * Dsz * 4);
    float*  h_f   = (float*) take((size_t)BTsz * Dsz * 4);
    half_t* x_h   = (half_t*)take((size_t)BTsz * Dsz * 2);
    half_t* enc_h = (half_t*)take((size_t)Bsz * Ssz * Dsz * 2);
    half_t* q_h   = (half_t*)take((size_t)BTsz * Dsz * 2);   // [B,H,T,DH]
    half_t* k_h   = (half_t*)take((size_t)BTsz * Dsz * 2);
    half_t* v_h   = (half_t*)take((size_t)BTsz * Dsz * 2);
    half_t* ctx_h = (half_t*)take((size_t)BTsz * Dsz * 2);
    float*  sc    = (float*) take((size_t)Bsz * Hsz * Tsz * Ssz * 4);
    half_t* pr    = (half_t*)take((size_t)Bsz * Hsz * Tsz * Ssz * 2);
    half_t* ff_h  = (half_t*)take((size_t)BTsz * Fsz * 2);
    half_t* wpool = (half_t*)take((size_t)Dsz * Vsz * 2);    // reused weight pool (max = out_w)

    // -------- helpers
    auto conv = [&](const float* s, half_t* d, long n) {
        long g = (n + 255) / 256;
        if (g > 16384) g = 16384;
        f32_to_f16_kernel<<<dim3((unsigned)g), dim3(256), 0, stream>>>(s, d, n);
    };
    auto gemm = [&](const half_t* A, const half_t* B, const float* bias, void* C,
                    int M, int N, int K, long sA, long sB, long sC, int nb,
                    int tB, int mode, int relu, float scale) {
        dim3 grid((N + BN - 1) / BN, (M + BM - 1) / BM, nb);
        gemm_wmma_kernel<<<grid, dim3(256), 0, stream>>>(
            A, B, bias, C, M, N, K, sA, sB, sC, tB, mode, relu, scale);
    };

    // -------- embedding + positional encoding
    embed_kernel<<<dim3(BTsz), dim3(256), 0, stream>>>(in_ids, emb, pos, x_f, x_h);
    conv(enc, enc_h, (long)Bsz * Ssz * Dsz);

    const long hstride = (long)Tsz * DHsz;   // per-(b,h) stride in head layout
    const long sstride = (long)Tsz * Ssz;    // per-(b,h) stride in score layout

    for (int l = 0; l < Lsz; ++l) {
        const float* g_ = ln_g + (long)l * Dsz;
        const float* b_ = ln_b + (long)l * Dsz;
        const long wofs = (long)l * Dsz * Dsz;

        // ================= self-attention =================
        conv(sa_w[0] + wofs, wpool, (long)Dsz * Dsz);
        gemm(x_h, wpool, sa_b[0] + l * Dsz, q_h, BTsz, Dsz, Dsz, 0, 0, 0, 1, 0, 2, 0, 1.0f);
        conv(sa_w[1] + wofs, wpool, (long)Dsz * Dsz);
        gemm(x_h, wpool, sa_b[1] + l * Dsz, k_h, BTsz, Dsz, Dsz, 0, 0, 0, 1, 0, 2, 0, 1.0f);
        conv(sa_w[2] + wofs, wpool, (long)Dsz * Dsz);
        gemm(x_h, wpool, sa_b[2] + l * Dsz, v_h, BTsz, Dsz, Dsz, 0, 0, 0, 1, 0, 2, 0, 1.0f);

        // scores = Q K^T / sqrt(DH), batched over B*H (transB: K stored N x K per head)
        gemm(q_h, k_h, nullptr, sc, Tsz, Ssz, DHsz, hstride, hstride, sstride,
             Bsz * Hsz, 1, 0, 0, 0.125f);
        softmax_kernel<<<dim3(Bsz * Hsz * Tsz), dim3(256), 0, stream>>>(sc, pr, 1);
        // ctx = P V (head-merge output back to [B*T, D])
        gemm(pr, v_h, nullptr, ctx_h, Tsz, DHsz, Ssz, sstride, hstride, 0,
             Bsz * Hsz, 0, 3, 0, 1.0f);

        conv(sa_w[3] + wofs, wpool, (long)Dsz * Dsz);
        gemm(ctx_h, wpool, sa_b[3] + l * Dsz, h_f, BTsz, Dsz, Dsz, 0, 0, 0, 1, 0, 0, 0, 1.0f);
        add_ln_kernel<<<dim3(BTsz), dim3(256), 0, stream>>>(x_f, h_f, g_, b_, x_f, x_h);

        // ================= cross-attention =================
        conv(ca_w[0] + wofs, wpool, (long)Dsz * Dsz);
        gemm(x_h, wpool, ca_b[0] + l * Dsz, q_h, BTsz, Dsz, Dsz, 0, 0, 0, 1, 0, 2, 0, 1.0f);
        conv(ca_w[1] + wofs, wpool, (long)Dsz * Dsz);
        gemm(enc_h, wpool, ca_b[1] + l * Dsz, k_h, Bsz * Ssz, Dsz, Dsz, 0, 0, 0, 1, 0, 2, 0, 1.0f);
        conv(ca_w[2] + wofs, wpool, (long)Dsz * Dsz);
        gemm(enc_h, wpool, ca_b[2] + l * Dsz, v_h, Bsz * Ssz, Dsz, Dsz, 0, 0, 0, 1, 0, 2, 0, 1.0f);

        gemm(q_h, k_h, nullptr, sc, Tsz, Ssz, DHsz, hstride, hstride, sstride,
             Bsz * Hsz, 1, 0, 0, 0.125f);
        softmax_kernel<<<dim3(Bsz * Hsz * Tsz), dim3(256), 0, stream>>>(sc, pr, 0);
        gemm(pr, v_h, nullptr, ctx_h, Tsz, DHsz, Ssz, sstride, hstride, 0,
             Bsz * Hsz, 0, 3, 0, 1.0f);

        conv(ca_w[3] + wofs, wpool, (long)Dsz * Dsz);
        gemm(ctx_h, wpool, ca_b[3] + l * Dsz, h_f, BTsz, Dsz, Dsz, 0, 0, 0, 1, 0, 0, 0, 1.0f);
        add_ln_kernel<<<dim3(BTsz), dim3(256), 0, stream>>>(x_f, h_f, g_, b_, x_f, x_h);

        // ================= FFN =================
        conv(ffn1_w + (long)l * Dsz * Fsz, wpool, (long)Dsz * Fsz);
        gemm(x_h, wpool, ffn1_b + (long)l * Fsz, ff_h, BTsz, Fsz, Dsz, 0, 0, 0, 1, 0, 1, 1, 1.0f);
        conv(ffn2_w + (long)l * Fsz * Dsz, wpool, (long)Fsz * Dsz);
        gemm(ff_h, wpool, ffn2_b + (long)l * Dsz, h_f, BTsz, Dsz, Fsz, 0, 0, 0, 1, 0, 0, 0, 1.0f);
        add_ln_kernel<<<dim3(BTsz), dim3(256), 0, stream>>>(x_f, h_f, g_, b_, x_f, x_h);
    }

    // -------- logits + log-softmax (in place in d_out)
    conv(out_w, wpool, (long)Dsz * Vsz);
    gemm(x_h, wpool, nullptr, (float*)d_out, BTsz, Vsz, Dsz, 0, 0, 0, 1, 0, 0, 0, 1.0f);
    log_softmax_kernel<<<dim3(BTsz), dim3(256), 0, stream>>>((float*)d_out);
}
